// DAWNBlock_75007308857725
// MI455X (gfx1250) — matrix operand order repaired and checked by reading, not compiled, not run on hardware
//
#include <hip/hip_runtime.h>
#include <hip/hip_bf16.h>

// ---------------------------------------------------------------------------
// Types for CDNA5 WMMA
// ---------------------------------------------------------------------------
typedef __bf16 v16bf __attribute__((ext_vector_type(16)));
typedef float  v8f   __attribute__((ext_vector_type(8)));

__device__ __forceinline__ v8f wmma_bf16(v16bf a, v16bf b, v8f c) {
    // 8 args: (neg_a, A, neg_b, B, c_mod, C, reuse_a, reuse_b)
    return __builtin_amdgcn_wmma_f32_16x16x32_bf16(false, a, false, b,
                                                   (short)0, c, false, false);
}

__device__ __forceinline__ unsigned short f2bf(float x) {
    unsigned int u = __float_as_uint(x);
    u += 0x7FFFu + ((u >> 16) & 1u);          // round-to-nearest-even
    return (unsigned short)(u >> 16);
}
__device__ __forceinline__ float bf2f(unsigned short h) {
    return __uint_as_float(((unsigned int)h) << 16);
}
__device__ __forceinline__ float gelu_erf(float x) {
    return 0.5f * x * (1.0f + erff(x * 0.70710678118654752f));
}

// CDNA5 async copy: per-lane 16B global -> LDS, tracked by ASYNCcnt.
__device__ __forceinline__ void async_cp_b128(unsigned lds_byte_off, const void* gptr) {
    asm volatile("global_load_async_to_lds_b128 %0, %1, off"
                 :: "v"(lds_byte_off), "v"(gptr) : "memory");
}
__device__ __forceinline__ void async_wait0() {
    asm volatile("s_wait_asynccnt 0x0" ::: "memory");
}
__device__ __forceinline__ void async_wait4() {
    asm volatile("s_wait_asynccnt 0x4" ::: "memory");
}

// ---------------------------------------------------------------------------
// Batched bf16 GEMM with WMMA:  C = act( A @ W^T + bias )
//   A : [M,K] bf16 row-major (lda, batch stride sA)
//   W : [N,K] bf16 row-major (ldb, batch stride sB)   (torch Linear layout)
//   C : optional f32 out (ldc) and/or bf16 out (ldcb; storeTrans=1 -> C^T)
// Block tile 128(M) x 128(N), K-step 32. 256 threads = 8 waves (wave32),
// waves 4(M) x 2(N); each wave 32x64 = 2x4 WMMA tiles -> 8 WMMA / K-step.
// Double-buffered LDS; tiles staged with GLOBAL_LOAD_ASYNC_TO_LDS_B128 so the
// next tile's copies overlap the current tile's WMMAs (ASYNCcnt in-order).
// Requires M%128==0, N%128==0, K%32==0, lda/ldb multiples of 8 (16B align).
// ---------------------------------------------------------------------------
__global__ __launch_bounds__(256)
void gemm_bf16_wmma(const unsigned short* __restrict__ A, int lda, long long sA,
                    const unsigned short* __restrict__ Wt, int ldb, long long sB,
                    const float* __restrict__ bias,
                    float* __restrict__ Cf, long long sCf, int ldc,
                    unsigned short* __restrict__ Cb, long long sCb, int ldcb,
                    int storeTrans,
                    int M, int N, int K, int act) {
    // packed bf16 pairs, row stride 20 uints = 80B (16B-aligned, bank-spread)
    __shared__ unsigned int Asu[2][128][20];   // A rows [m][K/2]
    __shared__ unsigned int Bsu[2][128][20];   // W rows [n][K/2]

    const int z = blockIdx.z;
    A += (size_t)z * sA;
    Wt += (size_t)z * sB;
    if (Cf) Cf += (size_t)z * sCf;
    if (Cb) Cb += (size_t)z * sCb;

    const int m0 = blockIdx.y * 128;
    const int n0 = blockIdx.x * 128;
    const int tid = threadIdx.x;
    const int lane = tid & 31;
    const int wm = ((tid >> 5) & 3) * 32;   // wave M offset
    const int wn = ((tid >> 5) >> 2) * 64;  // wave N offset

    v8f acc[2][4];
    #pragma unroll
    for (int i = 0; i < 2; ++i)
        #pragma unroll
        for (int j = 0; j < 4; ++j)
            #pragma unroll
            for (int r = 0; r < 8; ++r) acc[i][j][r] = 0.0f;

    const int rA = lane & 15;   // row within 16-row subtile
    const int hA = lane >> 4;   // K-half selector

    // per-thread staging coords: exactly 2 chunks per matrix per tile
    // chunk t: row = t>>2, col-chunk = t&3 ; threads handle t = tid, tid+256
    const int sr = tid >> 2;      // 0..63
    const int sc = tid & 3;       // 0..3  ((tid+256)&3 == tid&3)

    auto stage = [&](int bb, int kk) {
        const unsigned aB = (unsigned)(size_t)(&Asu[bb][0][0]);
        const unsigned bB = (unsigned)(size_t)(&Bsu[bb][0][0]);
        async_cp_b128(aB + (unsigned)(sr * 80 + sc * 16),
                      A + (size_t)(m0 + sr) * lda + kk + sc * 8);
        async_cp_b128(aB + (unsigned)((sr + 64) * 80 + sc * 16),
                      A + (size_t)(m0 + sr + 64) * lda + kk + sc * 8);
        async_cp_b128(bB + (unsigned)(sr * 80 + sc * 16),
                      Wt + (size_t)(n0 + sr) * ldb + kk + sc * 8);
        async_cp_b128(bB + (unsigned)((sr + 64) * 80 + sc * 16),
                      Wt + (size_t)(n0 + sr + 64) * ldb + kk + sc * 8);
    };

    union Frag { v16bf v; uint4 q[2]; };

    stage(0, 0);
    int buf = 0;
    for (int kk = 0; kk < K; kk += 32) {
        const bool more = (kk + 32) < K;
        if (more) {
            stage(buf ^ 1, kk + 32);   // overlap next tile's copy with compute
            async_wait4();             // tile kk's 4 per-thread copies done
        } else {
            async_wait0();
        }
        __syncthreads();

        // ---- fragments (contiguous k = 16*half + s layout): two b128 LDS loads each
        Frag a0, a1, b0, b1, b2, b3;
        a0.q[0] = *(const uint4*)&Asu[buf][wm + rA][8 * hA];
        a0.q[1] = *(const uint4*)&Asu[buf][wm + rA][8 * hA + 4];
        a1.q[0] = *(const uint4*)&Asu[buf][wm + 16 + rA][8 * hA];
        a1.q[1] = *(const uint4*)&Asu[buf][wm + 16 + rA][8 * hA + 4];
        b0.q[0] = *(const uint4*)&Bsu[buf][wn + rA][8 * hA];
        b0.q[1] = *(const uint4*)&Bsu[buf][wn + rA][8 * hA + 4];
        b1.q[0] = *(const uint4*)&Bsu[buf][wn + 16 + rA][8 * hA];
        b1.q[1] = *(const uint4*)&Bsu[buf][wn + 16 + rA][8 * hA + 4];
        b2.q[0] = *(const uint4*)&Bsu[buf][wn + 32 + rA][8 * hA];
        b2.q[1] = *(const uint4*)&Bsu[buf][wn + 32 + rA][8 * hA + 4];
        b3.q[0] = *(const uint4*)&Bsu[buf][wn + 48 + rA][8 * hA];
        b3.q[1] = *(const uint4*)&Bsu[buf][wn + 48 + rA][8 * hA + 4];

        acc[0][0] = wmma_bf16(a0.v, b0.v, acc[0][0]);
        acc[0][1] = wmma_bf16(a0.v, b1.v, acc[0][1]);
        acc[0][2] = wmma_bf16(a0.v, b2.v, acc[0][2]);
        acc[0][3] = wmma_bf16(a0.v, b3.v, acc[0][3]);
        acc[1][0] = wmma_bf16(a1.v, b0.v, acc[1][0]);
        acc[1][1] = wmma_bf16(a1.v, b1.v, acc[1][1]);
        acc[1][2] = wmma_bf16(a1.v, b2.v, acc[1][2]);
        acc[1][3] = wmma_bf16(a1.v, b3.v, acc[1][3]);
        __syncthreads();   // all waves done reading buf before it is re-staged
        buf ^= 1;
    }

    // ---- epilog: C 16x16 f32 layout (VGPR r: lanes0-15 M=r, lanes16-31 M=8+r)
    #pragma unroll
    for (int i = 0; i < 2; ++i) {
        #pragma unroll
        for (int j = 0; j < 4; ++j) {
            int col = n0 + wn + 16 * j + (lane & 15);
            int rbase = m0 + wm + 16 * i + 8 * (lane >> 4);
            float bv = bias ? bias[col] : 0.0f;
            #pragma unroll
            for (int r = 0; r < 8; ++r) {
                float c = acc[i][j][r] + bv;
                if (act == 1) c = gelu_erf(c);
                if (Cf) Cf[(size_t)(rbase + r) * ldc + col] = c;
                if (Cb) {
                    if (storeTrans)
                        Cb[(size_t)col * ldcb + (rbase + r)] = f2bf(c);
                    else
                        Cb[(size_t)(rbase + r) * ldcb + col] = f2bf(c);
                }
            }
        }
    }
}

// ---------------------------------------------------------------------------
// Elementwise / reduction helper kernels
// ---------------------------------------------------------------------------
__global__ void cvt_f32_bf16(const float* __restrict__ in,
                             unsigned short* __restrict__ out, long long n) {
    long long i = (long long)blockIdx.x * 256 + threadIdx.x;
    if (i < n) out[i] = f2bf(in[i]);
}

// one block per row: softmax(in[row,:]*scale) -> bf16
__global__ __launch_bounds__(256)
void softmax_rows(const float* __restrict__ in, unsigned short* __restrict__ out,
                  int NC, float scale) {
    __shared__ float red[256];
    const int row = blockIdx.x, tid = threadIdx.x;
    const float* r = in + (size_t)row * NC;
    unsigned short* o = out + (size_t)row * NC;
    float m = -3.4e38f;
    for (int c = tid; c < NC; c += 256) m = fmaxf(m, r[c] * scale);
    red[tid] = m; __syncthreads();
    for (int s = 128; s > 0; s >>= 1) {
        if (tid < s) red[tid] = fmaxf(red[tid], red[tid + s]);
        __syncthreads();
    }
    m = red[0]; __syncthreads();
    float sum = 0.0f;
    for (int c = tid; c < NC; c += 256) sum += expf(r[c] * scale - m);
    red[tid] = sum; __syncthreads();
    for (int s = 128; s > 0; s >>= 1) {
        if (tid < s) red[tid] += red[tid + s];
        __syncthreads();
    }
    float inv = 1.0f / red[0];
    for (int c = tid; c < NC; c += 256) o[c] = f2bf(expf(r[c] * scale - m) * inv);
}

// scores[b,n] = max_s aff[b,s,n]
__global__ void maxpool_seq(const float* __restrict__ aff, float* __restrict__ out,
                            int S, int NC, int total) {
    int t = blockIdx.x * 256 + threadIdx.x;
    if (t >= total) return;
    int b = t / NC, n = t % NC;
    const float* p = aff + (size_t)b * S * NC + n;
    float m = -3.4e38f;
    for (int s = 0; s < S; ++s) m = fmaxf(m, p[(size_t)s * NC]);
    out[t] = m;
}

// ps[b,n] = mean_s pa[b,s,n] (bf16 input)
__global__ void mean_seq_bf16(const unsigned short* __restrict__ in,
                              float* __restrict__ out, int S, int NC, int total) {
    int t = blockIdx.x * 256 + threadIdx.x;
    if (t >= total) return;
    int b = t / NC, n = t % NC;
    const unsigned short* p = in + (size_t)b * S * NC + n;
    float s = 0.0f;
    for (int i = 0; i < S; ++i) s += bf2f(p[(size_t)i * NC]);
    out[t] = s / (float)S;
}

// rank-selection top-k (set semantics, order-free): one block of N threads per batch
__global__ void topk_rank(const float* __restrict__ scores, int* __restrict__ idx,
                          int N, int k) {
    extern __shared__ float sh[];
    const int b = blockIdx.x, i = threadIdx.x;
    float v = scores[(size_t)b * N + i];
    sh[i] = v; __syncthreads();
    int rank = 0;
    for (int j = 0; j < N; ++j) {
        float u = sh[j];
        rank += (u > v) || (u == v && j < i);
    }
    if (rank < k) idx[(size_t)b * k + rank] = i;
}

// residual + LayerNorm -> bf16 (one block per row)
__global__ __launch_bounds__(256)
void ln_resid(const float* __restrict__ a, const float* __restrict__ r,
              const float* __restrict__ g, const float* __restrict__ bb,
              unsigned short* __restrict__ out, int NC) {
    __shared__ float red[256];
    const int row = blockIdx.x, tid = threadIdx.x;
    const float* pa = a + (size_t)row * NC;
    const float* pr = r + (size_t)row * NC;
    float s = 0.0f;
    for (int c = tid; c < NC; c += 256) s += pa[c] + pr[c];
    red[tid] = s; __syncthreads();
    for (int st = 128; st > 0; st >>= 1) {
        if (tid < st) red[tid] += red[tid + st];
        __syncthreads();
    }
    float mu = red[0] / (float)NC; __syncthreads();
    float vs = 0.0f;
    for (int c = tid; c < NC; c += 256) {
        float d = pa[c] + pr[c] - mu; vs += d * d;
    }
    red[tid] = vs; __syncthreads();
    for (int st = 128; st > 0; st >>= 1) {
        if (tid < st) red[tid] += red[tid + st];
        __syncthreads();
    }
    float inv = rsqrtf(red[0] / (float)NC + 1e-5f);
    unsigned short* o = out + (size_t)row * NC;
    for (int c = tid; c < NC; c += 256)
        o[c] = f2bf((pa[c] + pr[c] - mu) * inv * g[c] + bb[c]);
}

// out[b,s,k] = in[b,s, idx[b,k]]   (bf16 gather over columns)
__global__ void gather_cols(const unsigned short* __restrict__ in, int ncols,
                            const int* __restrict__ idx, int ktot,
                            unsigned short* __restrict__ out, int S, long long total) {
    long long t = (long long)blockIdx.x * 256 + threadIdx.x;
    if (t >= total) return;
    int k = (int)(t % ktot);
    long long bs = t / ktot;
    int b = (int)(bs / S);
    out[t] = in[bs * ncols + idx[(size_t)b * ktot + k]];
}

// out[b,p,k] = w[p, idx[b,k]]
__global__ void gather_w_cols(const unsigned short* __restrict__ w, int ncols,
                              const int* __restrict__ idx, int ktot, int NP,
                              unsigned short* __restrict__ out, long long total) {
    long long t = (long long)blockIdx.x * 256 + threadIdx.x;
    if (t >= total) return;
    int k = (int)(t % ktot);
    long long bp = t / ktot;
    int p = (int)(bp % NP);
    int b = (int)(bp / NP);
    out[t] = w[(size_t)p * ncols + idx[(size_t)b * ktot + k]];
}

// transposed row gather: out[b,d,k] = src[idx[b,k], d]   (out is [B,D,ktot])
__global__ void gather_rows_t(const unsigned short* __restrict__ src, int D,
                              const int* __restrict__ idx, int ktot,
                              unsigned short* __restrict__ out, long long total) {
    long long t = (long long)blockIdx.x * 256 + threadIdx.x;
    if (t >= total) return;
    int k = (int)(t % ktot);
    long long bd = t / ktot;
    int d = (int)(bd % D);
    int b = (int)(bd / D);
    out[t] = src[(size_t)idx[(size_t)b * ktot + k] * D + d];
}

// ---------------------------------------------------------------------------
// Host-side orchestration
// ---------------------------------------------------------------------------
static inline void* ws_alloc(char* base, size_t& off, size_t bytes) {
    void* p = base + off;
    off = (off + bytes + 255) & ~(size_t)255;
    return p;
}

extern "C" void kernel_launch(void* const* d_in, const int* in_sizes, int n_in,
                              void* d_out, int out_size, void* d_ws, size_t ws_size,
                              hipStream_t stream) {
    (void)in_sizes; (void)n_in; (void)out_size; (void)ws_size;
    const int B = 4, S = 1024, D = 1024, NI = 512, NP = 1024;
    const int KI = 256, KP = 512, HR = 8, HI = 4, DH = 128;
    const int M = B * S;
    const float SC = 0.08838834764831845f;  // 1/sqrt(128)

    const float* x      = (const float*)d_in[0];
    const float* r_wq   = (const float*)d_in[1];
    const float* r_wk   = (const float*)d_in[2];
    const float* r_wv   = (const float*)d_in[3];
    const float* r_bq   = (const float*)d_in[4];
    const float* r_bk   = (const float*)d_in[5];
    const float* r_bv   = (const float*)d_in[6];
    const float* r_wo   = (const float*)d_in[7];
    const float* r_bo   = (const float*)d_in[8];
    const float* aff_w  = (const float*)d_in[9];
    const float* aff_b  = (const float*)d_in[10];
    const float* pat    = (const float*)d_in[11];
    const float* i_wq   = (const float*)d_in[12];
    const float* i_wk   = (const float*)d_in[13];
    const float* i_wv   = (const float*)d_in[14];
    const float* i_bq   = (const float*)d_in[15];
    const float* i_bk   = (const float*)d_in[16];
    const float* i_bv   = (const float*)d_in[17];
    const float* i_wo   = (const float*)d_in[18];
    const float* i_bo   = (const float*)d_in[19];
    const float* ln_g   = (const float*)d_in[20];
    const float* ln_b   = (const float*)d_in[21];
    const float* comb_w = (const float*)d_in[22];
    const float* proj_w = (const float*)d_in[23];
    float* out = (float*)d_out;

    char* W = (char*)d_ws; size_t off = 0;
    typedef unsigned short u16;
    u16* xb   = (u16*)ws_alloc(W, off, (size_t)M * D * 2);
    u16* wrq  = (u16*)ws_alloc(W, off, (size_t)D * D * 2);
    u16* wrk  = (u16*)ws_alloc(W, off, (size_t)D * D * 2);
    u16* wrv  = (u16*)ws_alloc(W, off, (size_t)D * D * 2);
    u16* wro  = (u16*)ws_alloc(W, off, (size_t)D * D * 2);
    u16* waff = (u16*)ws_alloc(W, off, (size_t)NI * D * 2);
    u16* wpat = (u16*)ws_alloc(W, off, (size_t)NI * D * 2);
    u16* wiq  = (u16*)ws_alloc(W, off, (size_t)NI * NI * 2);
    u16* wik  = (u16*)ws_alloc(W, off, (size_t)NI * NI * 2);
    u16* wiv  = (u16*)ws_alloc(W, off, (size_t)NI * NI * 2);
    u16* wio  = (u16*)ws_alloc(W, off, (size_t)NI * NI * 2);
    u16* wcmb = (u16*)ws_alloc(W, off, (size_t)NP * NI * 2);
    u16* wprj = (u16*)ws_alloc(W, off, (size_t)NP * D * 2);
    u16* Qr   = (u16*)ws_alloc(W, off, (size_t)M * D * 2);
    u16* Kr   = (u16*)ws_alloc(W, off, (size_t)M * D * 2);
    u16* VTr  = (u16*)ws_alloc(W, off, (size_t)M * D * 2);   // [B][D][S]
    u16* Or   = (u16*)ws_alloc(W, off, (size_t)M * D * 2);
    u16* Ctx  = (u16*)ws_alloc(W, off, (size_t)M * D * 2);
    u16* Actb = (u16*)ws_alloc(W, off, (size_t)M * NI * 2);
    u16* Qi   = (u16*)ws_alloc(W, off, (size_t)M * NI * 2);
    u16* Ki   = (u16*)ws_alloc(W, off, (size_t)M * NI * 2);
    u16* VTi  = (u16*)ws_alloc(W, off, (size_t)M * NI * 2);  // [B][NI][S]
    u16* Oi   = (u16*)ws_alloc(W, off, (size_t)M * NI * 2);
    u16* Lnb  = (u16*)ws_alloc(W, off, (size_t)M * NI * 2);
    u16* attnB= (u16*)ws_alloc(W, off, (size_t)HR * S * S * 2);
    u16* selb = (u16*)ws_alloc(W, off, (size_t)B * S * KI * 2);
    u16* selw = (u16*)ws_alloc(W, off, (size_t)B * NP * KI * 2);
    u16* pab  = (u16*)ws_alloc(W, off, (size_t)B * S * NP * 2);
    u16* spab = (u16*)ws_alloc(W, off, (size_t)B * S * KP * 2);
    u16* sprjT= (u16*)ws_alloc(W, off, (size_t)B * D * KP * 2); // [B][D][KP]
    float* scF  = (float*)ws_alloc(W, off, (size_t)HR * S * S * 4);
    float* affF = (float*)ws_alloc(W, off, (size_t)M * NI * 4);
    float* actF = (float*)ws_alloc(W, off, (size_t)M * NI * 4);
    float* aoutF= (float*)ws_alloc(W, off, (size_t)M * NI * 4);
    float* rscF = (float*)ws_alloc(W, off, (size_t)B * NI * 4);
    float* psF  = (float*)ws_alloc(W, off, (size_t)B * NP * 4);
    int* idxI   = (int*)ws_alloc(W, off, (size_t)B * KI * 4);
    int* pidxI  = (int*)ws_alloc(W, off, (size_t)B * KP * 4);

    auto cvt = [&](const float* src, u16* dst, long long n) {
        cvt_f32_bf16<<<dim3((unsigned)((n + 255) / 256)), dim3(256), 0, stream>>>(src, dst, n);
    };
    cvt(x, xb, (long long)M * D);
    cvt(r_wq, wrq, (long long)D * D);  cvt(r_wk, wrk, (long long)D * D);
    cvt(r_wv, wrv, (long long)D * D);  cvt(r_wo, wro, (long long)D * D);
    cvt(aff_w, waff, (long long)NI * D); cvt(pat, wpat, (long long)NI * D);
    cvt(i_wq, wiq, (long long)NI * NI); cvt(i_wk, wik, (long long)NI * NI);
    cvt(i_wv, wiv, (long long)NI * NI); cvt(i_wo, wio, (long long)NI * NI);
    cvt(comb_w, wcmb, (long long)NP * NI); cvt(proj_w, wprj, (long long)NP * D);

    auto gemm = [&](const u16* A, int lda, long long sA,
                    const u16* Wm, int ldb, long long sB,
                    const float* bias,
                    float* Cf, long long sCf, int ldc,
                    u16* Cb, long long sCb, int ldcb, int storeTrans,
                    int Mm, int Nn, int Kk, int act, int batch) {
        dim3 g(Nn / 128, Mm / 128, batch);
        gemm_bf16_wmma<<<g, dim3(256), 0, stream>>>(A, lda, sA, Wm, ldb, sB,
                                                    bias, Cf, sCf, ldc,
                                                    Cb, sCb, ldcb, storeTrans,
                                                    Mm, Nn, Kk, act);
    };

    // ---- Router MHA ----
    gemm(xb, D, 0, wrq, D, 0, r_bq, nullptr, 0, 0, Qr, 0, D, 0, M, D, D, 0, 1);
    gemm(xb, D, 0, wrk, D, 0, r_bk, nullptr, 0, 0, Kr, 0, D, 0, M, D, D, 0, 1);
    // V projection stored transposed per batch: VTr[b][n][s]
    gemm(xb, D, (long long)S * D, wrv, D, 0, r_bv, nullptr, 0, 0,
         VTr, (long long)D * S, S, 1, S, D, D, 0, B);
    for (int b = 0; b < B; ++b) {
        const u16* Qb = Qr + (size_t)b * S * D;
        const u16* Kb = Kr + (size_t)b * S * D;
        const u16* Vb = VTr + (size_t)b * D * S;
        gemm(Qb, D, DH, Kb, D, DH, nullptr, scF, (long long)S * S, S,
             nullptr, 0, 0, 0, S, S, DH, 0, HR);
        softmax_rows<<<dim3(HR * S), dim3(256), 0, stream>>>(scF, attnB, S, SC);
        // attn @ V  ==  attn @ (V^T)^T : W rows are V^T rows h*DH..h*DH+127
        gemm(attnB, S, (long long)S * S, Vb, S, (long long)DH * S, nullptr,
             nullptr, 0, 0, Or + (size_t)b * S * D, DH, D, 0, S, DH, S, 0, HR);
    }
    gemm(Or, D, 0, wro, D, 0, r_bo, nullptr, 0, 0, Ctx, 0, D, 0, M, D, D, 0, 1);

    // ---- affinity + input-neuron activations ----
    gemm(Ctx, D, 0, waff, D, 0, aff_b, affF, 0, NI, nullptr, 0, 0, 0, M, NI, D, 0, 1);
    gemm(Ctx, D, 0, wpat, D, 0, nullptr, actF, 0, NI, Actb, 0, NI, 0, M, NI, D, 1, 1);

    // ---- Input MHA ----
    gemm(Actb, NI, 0, wiq, NI, 0, i_bq, nullptr, 0, 0, Qi, 0, NI, 0, M, NI, NI, 0, 1);
    gemm(Actb, NI, 0, wik, NI, 0, i_bk, nullptr, 0, 0, Ki, 0, NI, 0, M, NI, NI, 0, 1);
    gemm(Actb, NI, (long long)S * NI, wiv, NI, 0, i_bv, nullptr, 0, 0,
         VTi, (long long)NI * S, S, 1, S, NI, NI, 0, B);
    for (int b = 0; b < B; ++b) {
        const u16* Qb = Qi + (size_t)b * S * NI;
        const u16* Kb = Ki + (size_t)b * S * NI;
        const u16* Vb = VTi + (size_t)b * NI * S;
        gemm(Qb, NI, DH, Kb, NI, DH, nullptr, scF, (long long)S * S, S,
             nullptr, 0, 0, 0, S, S, DH, 0, HI);
        softmax_rows<<<dim3(HI * S), dim3(256), 0, stream>>>(scF, attnB, S, SC);
        gemm(attnB, S, (long long)S * S, Vb, S, (long long)DH * S, nullptr,
             nullptr, 0, 0, Oi + (size_t)b * S * NI, DH, NI, 0, S, DH, S, 0, HI);
    }
    gemm(Oi, NI, 0, wio, NI, 0, i_bo, aoutF, 0, NI, nullptr, 0, 0, 0, M, NI, NI, 0, 1);

    // ---- Router selection (weights == one_hot forward; order-free top-k) ----
    maxpool_seq<<<dim3((B * NI + 255) / 256), dim3(256), 0, stream>>>(affF, rscF, S, NI, B * NI);
    topk_rank<<<dim3(B), dim3(NI), NI * sizeof(float), stream>>>(rscF, idxI, NI, KI);

    // ---- Residual + LayerNorm ----
    ln_resid<<<dim3(M), dim3(256), 0, stream>>>(actF, aoutF, ln_g, ln_b, Lnb, NI);

    // ---- Gather selected input neurons + their comb_w columns ----
    long long tsel = (long long)B * S * KI;
    gather_cols<<<dim3((unsigned)((tsel + 255) / 256)), dim3(256), 0, stream>>>(
        Lnb, NI, idxI, KI, selb, S, tsel);
    long long tsw = (long long)B * NP * KI;
    gather_w_cols<<<dim3((unsigned)((tsw + 255) / 256)), dim3(256), 0, stream>>>(
        wcmb, NI, idxI, KI, NP, selw, tsw);

    // ---- ProcessNeurons: pa = gelu(sel @ sel_w^T) ----
    gemm(selb, KI, (long long)S * KI, selw, KI, (long long)NP * KI, nullptr,
         nullptr, 0, 0, pab, (long long)S * NP, NP, 0, S, NP, KI, 1, B);
    mean_seq_bf16<<<dim3((B * NP + 255) / 256), dim3(256), 0, stream>>>(pab, psF, S, NP, B * NP);
    topk_rank<<<dim3(B), dim3(NP), NP * sizeof(float), stream>>>(psF, pidxI, NP, KP);

    long long tspa = (long long)B * S * KP;
    gather_cols<<<dim3((unsigned)((tspa + 255) / 256)), dim3(256), 0, stream>>>(
        pab, NP, pidxI, KP, spab, S, tspa);
    // transposed projection gather: sprjT[b][d][k] = proj_w[pidx[b,k]][d]
    long long tprj = (long long)B * D * KP;
    gather_rows_t<<<dim3((unsigned)((tprj + 255) / 256)), dim3(256), 0, stream>>>(
        wprj, D, pidxI, KP, sprjT, tprj);

    // ---- Final projection: out = spa @ sprjT^T ----
    gemm(spab, KP, (long long)S * KP, sprjT, KP, (long long)D * KP, nullptr,
         out, (long long)S * D, D, nullptr, 0, 0, 0, S, D, KP, 0, B);
}
